// SelfRouting2d_83983790506265
// MI455X (gfx1250) — matrix operand order repaired and checked by reading, not compiled, not run on hardware
//
#include <hip/hip_runtime.h>
#include <hip/hip_bf16.h>

// CDNA5 / gfx1250 wave32 WMMA implementation of SelfRouting2d.
// d_in: a(4,32,32,32) f32, pose(4,512,32,32) f32, W1(288,128,16) f32,
//       W2(288,8,16) f32, b2(288,8) f32
// d_out: a_out(4,8,32,32) ++ pose_o(4,128,32,32), f32.

typedef __attribute__((ext_vector_type(16))) _Float16 v16h;
typedef __attribute__((ext_vector_type(8)))  float    v8f;

#define A_CAPS   32
#define B_CAPS   8
#define C_DIM    16
#define D_DIM    16
#define KKA      288       // 9 * 32
#define HW       32
#define TILE     4         // 4x4 pixel tile per workgroup
#define NPIX     16
#define HALO     6
#define KA_CHUNK 16
#define NCHUNK   (KKA / KA_CHUNK)   // 18 chunks; 8 WMMA K-steps each

__launch_bounds__(256)
__global__ void selfrouting2d_wmma(const float* __restrict__ act,
                                   const float* __restrict__ pose,
                                   const float* __restrict__ W1,
                                   const float* __restrict__ W2,
                                   const float* __restrict__ b2,
                                   float* __restrict__ out, int nb) {
    // LDS staging: pose halo tile (f16), per-chunk routing weights (f16),
    // running softmax-weighted sums (f32).
    __shared__ _Float16 poseT[HALO * HALO][A_CAPS * C_DIM];  // 36,864 B
    __shared__ _Float16 wbuf[NPIX][KA_CHUNK][B_CAPS];        //  4,096 B
    __shared__ float    arsum[NPIX][B_CAPS];                 //    512 B
    __shared__ float    ausum[NPIX];                         //     64 B

    const int tid  = threadIdx.x;
    const int b    = blockIdx.y;
    const int bx   = blockIdx.x & 7;   // 8 tiles per row (32 / 4)
    const int by   = blockIdx.x >> 3;

    if (tid < NPIX * B_CAPS) arsum[tid >> 3][tid & 7] = 0.0f;
    if (tid < NPIX)          ausum[tid] = 0.0f;

    // ---- Stage pose halo (6x6 positions x 512 channels) into LDS as f16 ----
    for (int i = tid; i < HALO * HALO * A_CAPS * C_DIM; i += 256) {
        const int pos = i >> 9;          // 0..35
        const int ch  = i & 511;         // acap*16 + c
        const int gy  = by * TILE - 1 + pos / HALO;
        const int gx  = bx * TILE - 1 + pos % HALO;
        float v = 0.0f;
        if (gy >= 0 && gy < HW && gx >= 0 && gx < HW)
            v = pose[(((b * 512 + ch) << 5) + gy) * HW + gx];
        poseT[pos][ch] = (_Float16)v;
    }
    __syncthreads();

    const int lane = tid & 31;
    const int wave = tid >> 5;           // wave == output capsule B
    const bool hi  = lane >= 16;

    v8f acc;
    #pragma unroll
    for (int j = 0; j < 8; ++j) acc[j] = 0.0f;

    // Phase-1 assignment: lane group of 16 pixels x 16 ka per chunk.
    const int pixP = tid & 15;
    const int kaL  = tid >> 4;
    const int py1  = pixP >> 2, px1 = pixP & 3;

    for (int chunk = 0; chunk < NCHUNK; ++chunk) {
        const int chunkBase = chunk * KA_CHUNK;

        // ---------- Phase 1: routing softmax -> w = au * r (per ka,pixel) ----------
        {
            const int ka   = chunkBase + kaL;
            const int kkv  = ka >> 5;            // kernel offset 0..8
            const int acap = ka & 31;            // input capsule
            const int ki = kkv / 3, kj = kkv % 3;
            const int pos = (py1 + ki) * HALO + (px1 + kj);
            const _Float16* pp = &poseT[pos][acap * C_DIM];

            float pu[C_DIM];
            #pragma unroll
            for (int c = 0; c < C_DIM; ++c) pu[c] = (float)pp[c];

            float lg[B_CAPS];
            const float* w2p = W2 + ka * (B_CAPS * C_DIM);
            #pragma unroll
            for (int e = 0; e < B_CAPS; ++e) {
                float s = b2[ka * B_CAPS + e];
                #pragma unroll
                for (int c = 0; c < C_DIM; ++c) s += w2p[e * C_DIM + c] * pu[c];
                lg[e] = s;
            }
            float m = lg[0];
            #pragma unroll
            for (int e = 1; e < B_CAPS; ++e) m = fmaxf(m, lg[e]);
            float ssum = 0.0f, r[B_CAPS];
            #pragma unroll
            for (int e = 0; e < B_CAPS; ++e) { r[e] = __expf(lg[e] - m); ssum += r[e]; }
            const float rinv = 1.0f / ssum;

            const int gy = by * TILE - 1 + py1 + ki;
            const int gx = bx * TILE - 1 + px1 + kj;
            float au = 0.0f;
            if (gy >= 0 && gy < HW && gx >= 0 && gx < HW)
                au = act[(((b * A_CAPS + acap) << 5) + gy) * HW + gx];

            #pragma unroll
            for (int e = 0; e < B_CAPS; ++e) {
                const float wv = au * r[e] * rinv;
                wbuf[pixP][kaL][e] = (_Float16)wv;
                atomicAdd(&arsum[pixP][e], wv);       // ds_add_f32
            }
            atomicAdd(&ausum[pixP], au);
        }
        __syncthreads();

        // ---------- Phase 2: vote GEMM, one wave per output capsule B ----------
        // D(16 x 16 pixels) += W1[B-block](16 x 32) @ (w .* pu)(32 x 16)
        const int d   = lane & 15;           // A-frag row (D index)
        const int cb  = hi ? 8 : 0;          // A-frag K sub-block
        const int pix = lane & 15;           // B-frag column (pixel)

        #pragma unroll
        for (int step = 0; step < 8; ++step) {
            const int ka0 = chunkBase + 2 * step;
            const int ka1 = ka0 + 1;

            // Prefetch next step's W1 slice into cache (global_prefetch_b8).
            const int kaN = (ka0 + 2 < KKA) ? (ka0 + 2) : ka0;
            __builtin_prefetch(W1 + ((kaN * 128 + wave * D_DIM + d) * C_DIM), 0, 1);

            // --- A fragment: W1 rows (o = B*16 + d) for ka0 | ka1, f32 -> f16 ---
            // lane<16: j=0..15 -> K=0..15 (ka0,c=j); lane>=16: K=16..31 (ka1,c=j)
            // (matches B fragment split).
            const float* w1p0 = W1 + (((hi ? ka1 : ka0) * 128 + wave * D_DIM + d) * C_DIM);
            const float* w1p1 = w1p0 + 8;
            v16h af;
            #pragma unroll
            for (int j = 0; j < 8; ++j) {
                af[j]     = (_Float16)w1p0[j];
                af[j + 8] = (_Float16)w1p1[j];
            }

            // --- B fragment: (w .* pu), K split ka0 (lanes 0-15) / ka1 (lanes 16-31) ---
            const int kasel = hi ? ka1 : ka0;
            const int kkv   = kasel >> 5;
            const int acap  = kasel & 31;
            const int pos   = ((pix >> 2) + kkv / 3) * HALO + ((pix & 3) + kkv % 3);
            const _Float16 wv = wbuf[pix][kasel - chunkBase][wave];
            const _Float16* pp = &poseT[pos][acap * C_DIM];
            v16h bf;
            #pragma unroll
            for (int j = 0; j < C_DIM; ++j) bf[j] = pp[j] * wv;

            acc = __builtin_amdgcn_wmma_f32_16x16x32_f16(
                false, af, false, bf, (short)0, acc, false, false);
        }
        __syncthreads();
    }

    // ---------------- Epilogue ----------------
    const int poseBase = nb * B_CAPS * HW * HW;   // a_out precedes pose_o

    // a_out[b, e, y, x] = ar_sum / au_sum
    if (tid < NPIX * B_CAPS) {
        const int pix = tid >> 3, e = tid & 7;
        const int lpix = (by * TILE + (pix >> 2)) * HW + (bx * TILE + (pix & 3));
        out[(b * B_CAPS + e) * (HW * HW) + lpix] = arsum[pix][e] / ausum[pix];
    }

    // pose_o[b, B*16+d, y, x] = acc / ar_sum[pix][B]
    {
        const int pix  = lane & 15;
        const int lpix = (by * TILE + (pix >> 2)) * HW + (bx * TILE + (pix & 3));
        const float inv = 1.0f / arsum[pix][wave];
        const int dbase = hi ? 8 : 0;
        #pragma unroll
        for (int j = 0; j < 8; ++j) {
            const int d = j + dbase;
            out[poseBase + ((b * 128 + wave * D_DIM + d) << 10) + lpix] = acc[j] * inv;
        }
    }
}

extern "C" void kernel_launch(void* const* d_in, const int* in_sizes, int n_in,
                              void* d_out, int out_size, void* d_ws, size_t ws_size,
                              hipStream_t stream) {
    (void)n_in; (void)out_size; (void)d_ws; (void)ws_size;
    const float* act  = (const float*)d_in[0];
    const float* pose = (const float*)d_in[1];
    const float* W1   = (const float*)d_in[2];
    const float* W2   = (const float*)d_in[3];
    const float* b2   = (const float*)d_in[4];
    float* out = (float*)d_out;
    const int nb = in_sizes[0] / (A_CAPS * HW * HW);   // batch (=4)

    dim3 grid((HW / TILE) * (HW / TILE), nb);          // 64 tiles x batch
    dim3 block(256);                                    // 8 wave32 = 8 output capsules
    hipLaunchKernelGGL(selfrouting2d_wmma, grid, block, 0, stream,
                       act, pose, W1, W2, b2, out, nb);
}
